// KNNClassifier_15908558864971
// MI455X (gfx1250) — matrix operand order repaired and checked by reading, not compiled, not run on hardware
//
#include <hip/hip_runtime.h>
#include <hip/hip_bf16.h>
#include <math.h>

// KNN classifier for MI455X (gfx1250, wave32, WMMA).
// Pipeline:
//   1) knn_cvt_mem : mem_features f32 -> f16 (halves streaming bytes; fits L2)
//   2) knn_prep_x  : x standardized + L2-normalized -> f16
//   3) knn_sims_topk: fused f16 WMMA GEMM, software-pipelined B operand
//                     (double-buffered fragments) + per-row running top-16.
//                     Single-wave workgroups: barriers replaced by s_wait_dscnt
//                     so next-tile loads overlap WMMA *and* the top-k scan.
//   4) knn_merge   : merge 32 slices x 16 candidates -> global top-16,
//                     softmax(tau), scatter into 1000-class logits, scale.

typedef _Float16 half_t;
typedef __attribute__((ext_vector_type(16))) _Float16 v16h;
typedef __attribute__((ext_vector_type(8)))  _Float16 v8h;
typedef __attribute__((ext_vector_type(4)))  _Float16 v4h;
typedef __attribute__((ext_vector_type(8)))  float    v8f;

namespace {
constexpr int   kB         = 2048;
constexpr int   kD         = 256;
constexpr int   kN         = 100000;
constexpr int   kK         = 16;
constexpr int   kC         = 1000;
constexpr float kTau       = 0.2f;
constexpr float kScale     = 20.0f;
constexpr int   kSlices    = 32;
constexpr int   kSliceCols = 3136;   // 196 tiles of 16; 32*3136 = 100352 >= kN
constexpr int   kTiles     = kSliceCols / 16;  // 196 (even -> clean 2x unroll)
constexpr float kNegInf    = -1.0e30f;
}

// Single-wave LDS fences (DS ops from one wave complete in order):
// store->load visibility: wait DScnt==0, plus compiler reordering fence.
__device__ __forceinline__ void wave_lds_store_fence() {
  asm volatile("s_wait_dscnt 0x0" ::: "memory");
}
// load->store direction is already ordered in the LDS pipeline per wave;
// only the compiler must not reorder. Zero instructions.
__device__ __forceinline__ void wave_compiler_fence() {
  asm volatile("" ::: "memory");
}

// ---------------------------------------------------------------- kernel 1 --
__global__ __launch_bounds__(256) void knn_cvt_mem(const float* __restrict__ mf,
                                                   half_t* __restrict__ mh) {
  const size_t i = (size_t)(blockIdx.x * blockDim.x + threadIdx.x) * 4;
  const float4 f = *(const float4*)(mf + i);
  v4h h;
  h[0] = (half_t)f.x; h[1] = (half_t)f.y; h[2] = (half_t)f.z; h[3] = (half_t)f.w;
  *(v4h*)(mh + i) = h;
}

// ---------------------------------------------------------------- kernel 2 --
__global__ __launch_bounds__(256) void knn_prep_x(const float* __restrict__ x,
                                                  const float* __restrict__ mean,
                                                  const float* __restrict__ stdv,
                                                  half_t* __restrict__ xh) {
  __shared__ float red[256];
  const int row = blockIdx.x;
  const int tid = threadIdx.x;              // tid == feature index (kD == 256)
  const float v = (x[(size_t)row * kD + tid] - mean[tid]) / stdv[tid];
  red[tid] = v * v;
  __syncthreads();
  for (int s = 128; s > 0; s >>= 1) {
    if (tid < s) red[tid] += red[tid + s];
    __syncthreads();
  }
  const float norm = fmaxf(sqrtf(red[0]), 1e-6f);
  xh[(size_t)row * kD + tid] = (half_t)(v / norm);
}

// ---------------------------------------------------------------- kernel 3 --
// One wave32 per (16-query block, N-slice). A fragments (16x256 f16) live in
// VGPRs for the whole slice. B operand: mem_features row-major [N,D] is
// column-major for B, so each lane's fragment slice is one contiguous 32B load.
// Software pipeline: issue next tile's 8 loads (into bn) before the 8 WMMAs
// consume the resident tile (bc); the loads then overlap WMMA + top-k scan.
__device__ __forceinline__ void knn_step(
    int n0, int next_n0, const half_t* __restrict__ mh,
    const v16h (&a)[8], const v16h (&bc)[8], v16h (&bn)[8],
    float (&tile)[16][17], float (&tv)[16][16], int (&ti)[16][16],
    float& minv, int& minp, int lane, int hi, int l15) {
  // ---- issue loads for the NEXT tile (consumed one step later)
  const int coln  = next_n0 + l15;
  const int colnc = coln < kN ? coln : (kN - 1);
  const half_t* bpn = mh + (size_t)colnc * kD + hi * 16;
#pragma unroll
  for (int kc = 0; kc < 8; ++kc) bn[kc] = *(const v16h*)(bpn + kc * 32);
  __builtin_prefetch(bpn + 16 * kD, 0, 1);   // tile t+2 into L2 (clamped addr)

  // ---- 8 WMMAs on the resident tile
  v8f c = {};
#pragma unroll
  for (int kc = 0; kc < 8; ++kc)
    c = __builtin_amdgcn_wmma_f32_16x16x32_f16(false, a[kc], false, bc[kc],
                                               (short)0, c, false, false);

  // C layout: VGPR v, lanes 0-15 -> (M=v, N=lane); lanes 16-31 -> (M=v+8).
#pragma unroll
  for (int v = 0; v < 8; ++v) tile[v + hi * 8][l15] = c[v];
  wave_lds_store_fence();                    // stores visible, no loadcnt drain

  // ---- top-k update: branch-free row-max prefilter, rare insertion path
  if (lane < 16) {
    const int row = lane;
    float m = tile[row][0];
#pragma unroll
    for (int c2 = 1; c2 < 16; ++c2) m = fmaxf(m, tile[row][c2]);
    if (m > minv) {
      for (int c2 = 0; c2 < 16; ++c2) {
        const int   n   = n0 + c2;
        const float val = tile[row][c2];
        if (n < kN && val > minv) {
          tv[row][minp] = val;
          ti[row][minp] = n;
          minv = tv[row][0]; minp = 0;       // rescan for new min
#pragma unroll
          for (int j = 1; j < 16; ++j)
            if (tv[row][j] < minv) { minv = tv[row][j]; minp = j; }
        }
      }
    }
  }
  wave_compiler_fence();   // same-wave DS pipeline orders load->store; no wait
}

__global__ __launch_bounds__(32) void knn_sims_topk(const half_t* __restrict__ xh,
                                                    const half_t* __restrict__ mh,
                                                    float* __restrict__ cval,
                                                    int*   __restrict__ cidx) {
  __shared__ float tile[16][17];   // 16x16 sims tile (+1 pad vs bank conflicts)
  __shared__ float tv[16][16];     // per-row top-16 values
  __shared__ int   ti[16][16];     // per-row top-16 global column indices

  const int lane = threadIdx.x;
  const int hi   = lane >> 4;      // 0: lanes 0-15, 1: lanes 16-31
  const int l15  = lane & 15;
  const int qb   = blockIdx.x;     // query block (16 rows)
  const int sl   = blockIdx.y;     // N slice

  // ---- A fragments: contiguous k order, matching the B loads.
  // lane<16 : elems 0..15 = K k0+0..15
  // lane>=16: elems 0..15 = K k0+16..31
  const half_t* xrow = xh + (size_t)(qb * 16 + l15) * kD;
  v16h a[8];
#pragma unroll
  for (int kc = 0; kc < 8; ++kc) {
    const int k0 = kc * 32;
    a[kc] = *(const v16h*)(xrow + k0 + hi * 16);
  }

  for (int i = lane; i < 256; i += 32) {
    ((float*)tv)[i] = kNegInf;
    ((int*)ti)[i]   = 0;
  }
  wave_lds_store_fence();

  float minv = kNegInf;   // valid in lanes 0-15 (row owners)
  int   minp = 0;

  const int nbase = sl * kSliceCols;

  // ---- prologue: load tile 0 into buffer A
  v16h bA[8], bB[8];
  {
    const int col  = nbase + l15;
    const int colc = col < kN ? col : (kN - 1);
    const half_t* bp = mh + (size_t)colc * kD + hi * 16;
#pragma unroll
    for (int kc = 0; kc < 8; ++kc) bA[kc] = *(const v16h*)(bp + kc * 32);
  }

  // ---- main loop, 2x unrolled so the double buffers alternate without copies
  for (int t = 0; t < kTiles; t += 2) {
    const int n0 = nbase + t * 16;
    const int n1 = n0 + 16;
    const int n2 = (t + 2 < kTiles) ? (n1 + 16) : n1;   // clamp: reload, unused
    knn_step(n0, n1, mh, a, bA, bB, tile, tv, ti, minv, minp, lane, hi, l15);
    knn_step(n1, n2, mh, a, bB, bA, tile, tv, ti, minv, minp, lane, hi, l15);
  }

  wave_lds_store_fence();   // tv/ti updates from lanes 0-15 visible to all lanes
  for (int i = lane; i < 256; i += 32) {
    const int row = i >> 4, j = i & 15;
    const size_t o = ((size_t)(qb * 16 + row) * kSlices + sl) * kK + j;
    cval[o] = tv[row][j];
    cidx[o] = ti[row][j];
  }
}

// ---------------------------------------------------------------- kernel 4 --
__global__ __launch_bounds__(128) void knn_merge(const float* __restrict__ cval,
                                                 const int*   __restrict__ cidx,
                                                 const int*   __restrict__ labels,
                                                 float* __restrict__ out) {
  constexpr int NC = kSlices * kK;          // 512 candidates per query
  __shared__ float cv[NC];
  __shared__ int   ci[NC];
  __shared__ float rv[128];
  __shared__ int   ri[128];
  __shared__ float bestv[kK];
  __shared__ int   besti[kK];

  const int tid = threadIdx.x;
  const int b   = blockIdx.x;

  for (int i = tid; i < NC; i += 128) {
    cv[i] = cval[(size_t)b * NC + i];
    ci[i] = cidx[(size_t)b * NC + i];
  }
  for (int i = tid; i < kC; i += 128) out[(size_t)b * kC + i] = 0.0f;
  __syncthreads();

  for (int r = 0; r < kK; ++r) {            // 16 rounds of parallel argmax
    float mv = -2.0e30f;
    int   mp = 0;
    for (int i = tid; i < NC; i += 128)
      if (cv[i] > mv) { mv = cv[i]; mp = i; }
    rv[tid] = mv; ri[tid] = mp;
    __syncthreads();
    for (int s = 64; s > 0; s >>= 1) {
      if (tid < s && rv[tid + s] > rv[tid]) { rv[tid] = rv[tid + s]; ri[tid] = ri[tid + s]; }
      __syncthreads();
    }
    if (tid == 0) {
      bestv[r] = rv[0];
      besti[r] = ci[ri[0]];
      cv[ri[0]] = -3.0e30f;                 // remove from further rounds
    }
    __syncthreads();
  }

  if (tid == 0) {                           // softmax over 16 + scatter (no races)
    const float m = bestv[0];               // round 0 extracted the max
    float w[kK], s = 0.0f;
    for (int j = 0; j < kK; ++j) { w[j] = expf((bestv[j] - m) / kTau); s += w[j]; }
    const float inv = kScale / s;
    float* row = out + (size_t)b * kC;
    for (int j = 0; j < kK; ++j) row[labels[besti[j]]] += w[j] * inv;
  }
}

// -------------------------------------------------------------------- host --
extern "C" void kernel_launch(void* const* d_in, const int* in_sizes, int n_in,
                              void* d_out, int out_size, void* d_ws, size_t ws_size,
                              hipStream_t stream) {
  const float* x      = (const float*)d_in[0];
  const float* mean   = (const float*)d_in[1];
  const float* stdv   = (const float*)d_in[2];
  const float* memf   = (const float*)d_in[3];
  const int*   labels = (const int*)d_in[4];
  float*       out    = (float*)d_out;
  (void)in_sizes; (void)n_in; (void)out_size; (void)ws_size;

  // Workspace layout (~60.7 MB total):
  char* ws = (char*)d_ws;
  size_t off = 0;
  half_t* mh   = (half_t*)(ws + off); off += (size_t)kN * kD * sizeof(half_t);            // 51.2 MB
  half_t* xh   = (half_t*)(ws + off); off += (size_t)kB * kD * sizeof(half_t);            //  1.0 MB
  float*  cval = (float*)(ws + off);  off += (size_t)kB * kSlices * kK * sizeof(float);   //  4.2 MB
  int*    cidx = (int*)(ws + off);    off += (size_t)kB * kSlices * kK * sizeof(int);     //  4.2 MB

  knn_cvt_mem  <<<kN * kD / 4 / 256, 256, 0, stream>>>(memf, mh);
  knn_prep_x   <<<kB, kD, 0, stream>>>(x, mean, stdv, xh);
  knn_sims_topk<<<dim3(kB / 16, kSlices), 32, 0, stream>>>(xh, mh, cval, cidx);
  knn_merge    <<<kB, 128, 0, stream>>>(cval, cidx, labels, out);
}
